// Generator_85048942395814
// MI455X (gfx1250) — matrix operand order repaired and checked by reading, not compiled, not run on hardware
//
#include <hip/hip_runtime.h>
#include <hip/hip_bf16.h>

// ---------------------------------------------------------------------------
// MI455X (gfx1250) implementation of the DGMG-style generator loss.
// Sequential RNN chain -> single persistent 512-thread workgroup (16 wave32
// waves) doing bf16 WMMA (v_wmma_f32_16x16x32_bf16) with f32 accumulation.
// Weights pre-packed to WMMA-fragment-major bf16 so every B-operand load is a
// coalesced 1KB wave read from L2 (all weights ~8MB, L2-resident).
// v2: B-tile reuse across batch tiles (each wave owns one hidden tile for all
// 4 batch tiles) + software-pipelined ping-pong B buffers so WMMA issue
// overlaps L2 load latency instead of s_wait_loadcnt 0x0 per tile.
// ---------------------------------------------------------------------------

typedef __bf16 bf16_t;
typedef __attribute__((ext_vector_type(16))) __bf16 v16bf;
typedef __attribute__((ext_vector_type(8)))  __bf16 v8bf;
typedef __attribute__((ext_vector_type(8)))  float  v8f;

#define S_  40
#define B_  64
#define H_  256
#define NE_ 128
#define EE_ 64

__device__ const int kNC[4] = {43, 7, 5, 4};
__device__ const int kEC[4] = {5, 4, 3, 3};

// ---------------------------------------------------------------------------
// WMMA helpers
// ---------------------------------------------------------------------------
__device__ __forceinline__ v8f wmma_bf16(v16bf a, v16bf b, v8f c) {
  return __builtin_amdgcn_wmma_f32_16x16x32_bf16(
      /*neg_a=*/false, a, /*neg_b=*/false, b,
      /*c_mod=*/(short)0, c, /*reuse_a=*/false, /*reuse_b=*/false);
}

// A fragment (16x32 bf16) from a row-major [rows][w] bf16 tensor.
// A layout: lanes 0-15 -> M=lane, K {0..15};
//           lanes 16-31 -> M=lane-16, K {16..31}.
__device__ __forceinline__ v16bf load_a(const bf16_t* __restrict__ t, int w,
                                        int m, int kt, int lane) {
  const bf16_t* p = t + (size_t)(16 * m + (lane & 15)) * w + 32 * kt +
                    ((lane >> 4) << 4);
  union { v16bf v; v8bf h[2]; } u;
  u.h[0] = *(const v8bf*)(p);
  u.h[1] = *(const v8bf*)(p + 8);
  return u.v;
}

// B fragment (32x16 bf16) from fragment-major blob: 512 bf16 per tile,
// lane's 16 elements contiguous at lane*16.
__device__ __forceinline__ v16bf load_b(const bf16_t* __restrict__ p) {
  union { v16bf v; v8bf h[2]; } u;
  u.h[0] = *(const v8bf*)(p);
  u.h[1] = *(const v8bf*)(p + 8);
  return u.v;
}

__device__ __forceinline__ float sigmoidf_(float x) {
  return 1.f / (1.f + __expf(-x));
}
__device__ __forceinline__ void sync_all() { __threadfence(); __syncthreads(); }

// ---------------------------------------------------------------------------
// One LSTM cell step: g = [x||h] @ [Wih||Whh]^T (+b), gates -> c2,h2.
// Gate-interleaved weight packing: global N-tile ntg = 4*ht + q so each wave
// (= one hidden tile ht) owns its i/f/g/o 16x16 tiles for ALL batch tiles and
// finishes the nonlinearity entirely in the WMMA accumulator layout.
// B tiles are loaded once per K-slab and consumed by 2 batch tiles; the next
// K-slab's B tiles are in flight while the current WMMAs execute.
// ---------------------------------------------------------------------------
__device__ __forceinline__ void lstm_cell(
    const bf16_t* __restrict__ x, int xw,
    const bf16_t* __restrict__ hprev,
    bf16_t* __restrict__ hout,
    float* __restrict__ cstate,
    const bf16_t* __restrict__ blob,
    const float* __restrict__ bias,
    int Kx, int tid) {
  const int lane = tid & 31;
  const int ht   = tid >> 5;            // wave id == hidden tile (16 waves)
  const int KxT  = Kx >> 5;
  const int KT   = KxT + 8;             // + H_/32  (always even: 12 or 16)
  const int nb   = 4 * ht;              // base gate tile (i,f,g,o)
  const bf16_t* bp0 = blob + ((size_t)nb << 9) + (lane << 4);

  auto loadA = [&](int m, int kt) -> v16bf {
    return (kt < KxT) ? load_a(x, xw, m, kt, lane)
                      : load_a(hprev, H_, m, kt - KxT, lane);
  };

  for (int mp = 0; mp < 2; ++mp) {      // batch tiles {0,1} then {2,3}
    const int m0 = 2 * mp, m1 = 2 * mp + 1;
    v8f acc0[4] = {}, acc1[4] = {};
    v16bf cur[4], nxt[4];
#pragma unroll
    for (int q = 0; q < 4; ++q) cur[q] = load_b(bp0 + (q << 9));

    for (int kt = 0; kt < KT; kt += 2) {
      // issue next slab's B loads before consuming the current slab
      const bf16_t* bpn = bp0 + (size_t)(kt + 1) * 32768;
#pragma unroll
      for (int q = 0; q < 4; ++q) nxt[q] = load_b(bpn + (q << 9));
      if (kt + 4 < KT)  // pre-warm L2->near for the slab after next
        __builtin_prefetch(bp0 + (size_t)(kt + 4) * 32768, 0, 1);
      v16bf a0 = loadA(m0, kt);
      v16bf a1 = loadA(m1, kt);
#pragma unroll
      for (int q = 0; q < 4; ++q) acc0[q] = wmma_bf16(a0, cur[q], acc0[q]);
#pragma unroll
      for (int q = 0; q < 4; ++q) acc1[q] = wmma_bf16(a1, cur[q], acc1[q]);

      if (kt + 2 < KT) {
        const bf16_t* bp2 = bp0 + (size_t)(kt + 2) * 32768;
#pragma unroll
        for (int q = 0; q < 4; ++q) cur[q] = load_b(bp2 + (q << 9));
      }
      a0 = loadA(m0, kt + 1);
      a1 = loadA(m1, kt + 1);
#pragma unroll
      for (int q = 0; q < 4; ++q) acc0[q] = wmma_bf16(a0, nxt[q], acc0[q]);
#pragma unroll
      for (int q = 0; q < 4; ++q) acc1[q] = wmma_bf16(a1, nxt[q], acc1[q]);
    }

    // ---- gates -> c2, h2, entirely in the WMMA C layout ----
    const int nc   = lane & 15;
    const int col  = ht * 16 + nc;
    const float bi = bias[(nb + 0) * 16 + nc];
    const float bf = bias[(nb + 1) * 16 + nc];
    const float bg = bias[(nb + 2) * 16 + nc];
    const float bo = bias[(nb + 3) * 16 + nc];
#pragma unroll
    for (int half = 0; half < 2; ++half) {
      v8f* ac = half ? acc1 : acc0;
      const int m = half ? m1 : m0;
      const int rbase = 16 * m + ((lane >> 4) << 3);
#pragma unroll
      for (int e = 0; e < 8; ++e) {
        const int row = rbase + e;      // WMMA C layout: M = e (+8 hi half)
        const float ig = sigmoidf_(ac[0][e] + bi);
        const float fg = sigmoidf_(ac[1][e] + bf);
        const float gg = tanhf(ac[2][e] + bg);
        const float og = sigmoidf_(ac[3][e] + bo);
        const float cp = cstate[row * H_ + col];
        const float c2 = fg * cp + ig * gg;
        cstate[row * H_ + col] = c2;
        hout[row * H_ + col]   = (bf16_t)(og * tanhf(c2));
      }
    }
  }
}

// Dense projection D = A(64xK) @ W^T (+b): WMMA with ping-pong B buffers,
// optional f32 LDS out + bf16 global out. K = KT*32 (even), N = Nt*16.
__device__ __forceinline__ void dense_out(
    const bf16_t* __restrict__ A, const bf16_t* __restrict__ blob,
    const float* __restrict__ bias, int Nt, int KT,
    float* __restrict__ sout, bf16_t* __restrict__ gout, int tid) {
  const int lane = tid & 31, wave = tid >> 5;
  const int N = Nt * 16;
  const size_t kstride = (size_t)Nt << 9;
  for (int u = wave; u < 4 * Nt; u += 16) {
    const int m = u & 3, nt = u >> 2;
    const bf16_t* base = blob + ((size_t)nt << 9) + (lane << 4);
    v8f acc = {};
    v16bf b0 = load_b(base), b1;
    for (int kt = 0; kt < KT; kt += 2) {
      b1 = load_b(base + (size_t)(kt + 1) * kstride);
      v16bf a = load_a(A, KT * 32, m, kt, lane);
      acc = wmma_bf16(a, b0, acc);
      if (kt + 2 < KT) b0 = load_b(base + (size_t)(kt + 2) * kstride);
      a = load_a(A, KT * 32, m, kt + 1, lane);
      acc = wmma_bf16(a, b1, acc);
    }
    const int col   = nt * 16 + (lane & 15);
    const float bb  = bias[col];
    const int rbase = 16 * m + ((lane >> 4) << 3);
#pragma unroll
    for (int e = 0; e < 8; ++e) {
      const float v = acc[e] + bb;
      if (sout) sout[(rbase + e) * N + col] = v;
      if (gout) gout[(rbase + e) * N + col] = (bf16_t)v;
    }
  }
}

// ---------------------------------------------------------------------------
// Weight packing kernels (f32 row-major -> bf16 WMMA-fragment-major)
// ---------------------------------------------------------------------------
__global__ void pack_lstm_w(const float* __restrict__ Wih,
                            const float* __restrict__ Whh,
                            bf16_t* __restrict__ blob, int Kx, int total) {
  int idx = blockIdx.x * 256 + threadIdx.x;
  if (idx >= total) return;
  const int e    = idx & 15;
  const int lane = (idx >> 4) & 31;
  const int ntg  = (idx >> 9) & 63;
  const int kt   = idx >> 15;
  const int k    = kt * 32 + ((lane >> 4) << 4) + e;
  const int nc   = lane & 15, ht = ntg >> 2, q = ntg & 3;
  const int n    = q * H_ + ht * 16 + nc;     // gate-interleaved permute
  const float w  = (k < Kx) ? Wih[n * Kx + k] : Whh[n * H_ + (k - Kx)];
  blob[idx] = (bf16_t)w;
}

__global__ void pack_lstm_bias(const float* __restrict__ bih,
                               const float* __restrict__ bhh,
                               float* __restrict__ bias_blob) {
  int idx = blockIdx.x * 256 + threadIdx.x;
  if (idx >= 1024) return;
  const int nc = idx & 15, ntg = idx >> 4;
  const int ht = ntg >> 2, q = ntg & 3;
  const int n  = q * H_ + ht * 16 + nc;
  bias_blob[idx] = bih[n] + bhh[n];
}

__global__ void pack_dense_w(const float* __restrict__ W,
                             bf16_t* __restrict__ blob, int Nt, int K,
                             int total) {
  int idx = blockIdx.x * 256 + threadIdx.x;
  if (idx >= total) return;
  const int e    = idx & 15;
  const int lane = (idx >> 4) & 31;
  const int nt   = (idx >> 9) % Nt;
  const int kt   = idx / (Nt << 9);
  const int k    = kt * 32 + ((lane >> 4) << 4) + e;
  const int n    = nt * 16 + (lane & 15);
  blob[idx] = (bf16_t)W[n * K + k];
}

__global__ void cvt_f32_bf16(const float* __restrict__ src,
                             bf16_t* __restrict__ dst, int n) {
  int i = blockIdx.x * 256 + threadIdx.x;
  if (i < n) dst[i] = (bf16_t)src[i];
}

// ---------------------------------------------------------------------------
// Persistent main kernel
// ---------------------------------------------------------------------------
struct RP {
  const bf16_t* WN[4];
  const bf16_t* WE[4];
  const bf16_t* WPROJ;
  const bf16_t* WNOUT;
  const bf16_t* WEOUT;
  const float* BN[4];
  const float* BE[4];
  const float* proj_b;
  const float* nout_b;
  const float* eout_b;
  const bf16_t* zbf;
  bf16_t* zc;     // 64*256
  bf16_t* embs;   // S*64*128
  bf16_t* hN;     // [4][2][64*256] double-buffered
  bf16_t* hE;     // [4][2][64*256]
  bf16_t* xnode;  // 64*256
  float* cN;      // [4][64*256]
  float* cE;      // [4][64*256]
  const int* atom;
  const int* bond;
  const float* ncW[4];
  const float* ncB[4];
  const float* ecW[4];
  const float* ecB[4];
  float* out;
};

__global__ __launch_bounds__(512, 1) void rnn_main(RP p) {
  const int tid = threadIdx.x;
  __shared__ float s_emb[B_ * NE_];
  __shared__ float s_red[512];
  __shared__ float s_acc[3];  // 0: node loss, 1: edge loss, 2: per-step edge sum
  if (tid < 3) s_acc[tid] = 0.f;
  __syncthreads();

  // zc = proj(z)
  dense_out(p.zbf, p.WPROJ, p.proj_b, /*Nt=*/16, /*KT=*/4, nullptr, p.zc, tid);
  sync_all();

  // set_context: h <- zc (all 4 layers), c <- 0 (memset on host side)
  for (int idx = tid; idx < B_ * H_; idx += 512) {
    bf16_t v = p.zc[idx];
    for (int l = 0; l < 4; ++l) p.hN[(l * 2 + 0) * (B_ * H_) + idx] = v;
  }
  sync_all();

  int pn[4] = {0, 0, 0, 0};
  for (int i = 0; i < S_; ++i) {
    // ---------------- node stack ----------------
    const bf16_t* xin = p.xnode;
    int xw = H_;
    for (int l = 0; l < 4; ++l) {
      lstm_cell(xin, xw, p.hN + (l * 2 + pn[l]) * (B_ * H_),
                p.hN + (l * 2 + (pn[l] ^ 1)) * (B_ * H_),
                p.cN + l * (B_ * H_), p.WN[l], p.BN[l], H_, tid);
      sync_all();
      pn[l] ^= 1;
      xin = p.hN + (l * 2 + pn[l]) * (B_ * H_);
      xw  = H_;
    }
    // node_emb = node_out(h3)
    dense_out(p.hN + (3 * 2 + pn[3]) * (B_ * H_), p.WNOUT, p.nout_b, 8, 8,
              s_emb, p.embs + (size_t)i * (B_ * NE_), tid);
    sync_all();

    // node classifier loss (skipped at i==0)
    float part = 0.f;
    if (tid < 256 && i > 0) {
      const int b = tid >> 2, j = tid & 3;
      const int C = kNC[j];
      const float* W = p.ncW[j];
      const float* bb = p.ncB[j];
      float logits[48];
      float mx = -3.4e38f;
      for (int c = 0; c < C; ++c) {
        float s = bb[c];
        for (int k = 0; k < NE_; ++k) s += W[c * NE_ + k] * s_emb[b * NE_ + k];
        logits[c] = s;
        mx = fmaxf(mx, s);
      }
      float lse = 0.f;
      for (int c = 0; c < C; ++c) lse += __expf(logits[c] - mx);
      lse = mx + __logf(lse);
      const int tg = p.atom[(i * B_ + b) * 4 + j];
      if (tg >= 0 && tg < C) part = lse - logits[tg];
    }
    s_red[tid] = part;
    __syncthreads();
    for (int s = 256; s >= 1; s >>= 1) {
      if (tid < s) s_red[tid] += s_red[tid + s];
      __syncthreads();
    }
    if (tid == 0 && i > 0) s_acc[0] += s_red[0] * (1.f / (float)B_);
    __syncthreads();

    // ---------------- edge scan ----------------
    const float* cN3 = p.cN + 3 * (B_ * H_);
    for (int idx = tid; idx < B_ * H_; idx += 512) {
      bf16_t hv = (bf16_t)cN3[idx];  // edge context = node last-layer c
      for (int l = 0; l < 4; ++l) {
        p.hE[(l * 2 + 0) * (B_ * H_) + idx] = hv;
        p.cE[l * (B_ * H_) + idx] = 0.f;
      }
    }
    if (tid == 0) s_acc[2] = 0.f;
    sync_all();

    int pe[4] = {0, 0, 0, 0};
    for (int t = 0; t < i; ++t) {       // only valid steps mutate state
      const bf16_t* exi = p.embs + (size_t)(i - 1 - t) * (B_ * NE_);
      int xwE = NE_;
      for (int l = 0; l < 4; ++l) {
        lstm_cell(exi, xwE, p.hE + (l * 2 + pe[l]) * (B_ * H_),
                  p.hE + (l * 2 + (pe[l] ^ 1)) * (B_ * H_),
                  p.cE + l * (B_ * H_), p.WE[l], p.BE[l],
                  l == 0 ? NE_ : H_, tid);
        sync_all();
        pe[l] ^= 1;
        exi = p.hE + (l * 2 + pe[l]) * (B_ * H_);
        xwE = H_;
      }
      dense_out(p.hE + (3 * 2 + pe[3]) * (B_ * H_), p.WEOUT, p.eout_b, 4, 8,
                s_emb, nullptr, tid);
      sync_all();

      float ep = 0.f;
      if (tid < 256) {
        const int b = tid >> 2, j = tid & 3;
        const int C = kEC[j];
        const float* W = p.ecW[j];
        const float* bb = p.ecB[j];
        float logits[8];
        float mx = -3.4e38f;
        for (int c = 0; c < C; ++c) {
          float s = bb[c];
          for (int k = 0; k < EE_; ++k) s += W[c * EE_ + k] * s_emb[b * EE_ + k];
          logits[c] = s;
          mx = fmaxf(mx, s);
        }
        float lse = 0.f;
        for (int c = 0; c < C; ++c) lse += __expf(logits[c] - mx);
        lse = mx + __logf(lse);
        const int src = i - 1 - t;
        const int tg = p.bond[((i * B_ + b) * (S_ - 1) + src) * 4 + j];
        if (tg >= 0 && tg < C) ep = lse - logits[tg];
      }
      s_red[tid] = ep;
      __syncthreads();
      for (int s = 256; s >= 1; s >>= 1) {
        if (tid < s) s_red[tid] += s_red[tid + s];
        __syncthreads();
      }
      if (tid == 0) s_acc[2] += s_red[0];
      __syncthreads();
    }
    if (tid == 0 && i > 0) s_acc[1] += s_acc[2] / (float)(i * B_);
    __syncthreads();

    // x_next = (i==0) ? node c3 : edge c3
    const float* xsel = (i == 0) ? (p.cN + 3 * (B_ * H_)) : (p.cE + 3 * (B_ * H_));
    for (int idx = tid; idx < B_ * H_; idx += 512)
      p.xnode[idx] = (bf16_t)xsel[idx];
    sync_all();
  }

  if (tid == 0)
    p.out[0] = s_acc[0] * (1.f / ((S_ - 1) * 4.f)) +
               s_acc[1] * (1.f / ((S_ - 1) * 4.f));
}

// ---------------------------------------------------------------------------
// Host launcher
// ---------------------------------------------------------------------------
extern "C" void kernel_launch(void* const* d_in, const int* in_sizes, int n_in,
                              void* d_out, int out_size, void* d_ws,
                              size_t ws_size, hipStream_t stream) {
  (void)in_sizes; (void)n_in; (void)out_size; (void)ws_size;
  // Input order (setup_inputs dict insertion order, params leaves in order):
  // 0:z 1:atom_target 2:bond_target 3:proj.W 4:proj.b
  // 5..20:  node_lstm[l]{Wih,Whh,bih,bhh}
  // 21:node_out.W 22:node_out.b
  // 23..38: edge_lstm[l]{Wih,Whh,bih,bhh}
  // 39:edge_out.W 40:edge_out.b
  // 41..48: node_cls[j]{W,b}   49..56: edge_cls[j]{W,b}
  auto F = [&](int i) { return (const float*)d_in[i]; };
  auto I = [&](int i) { return (const int*)d_in[i]; };

  char* ws = (char*)d_ws;
  size_t off = 0;
  auto carve = [&](size_t bytes) -> char* {
    char* p = ws + off;
    off = (off + bytes + 255) & ~(size_t)255;
    return p;
  };

  const int BH = B_ * H_;
  bf16_t* WN[4]; bf16_t* WE[4];
  float*  BN[4]; float*  BE[4];
  for (int l = 0; l < 4; ++l) WN[l] = (bf16_t*)carve(16 * 32768 * 2);
  WE[0] = (bf16_t*)carve(12 * 32768 * 2);
  for (int l = 1; l < 4; ++l) WE[l] = (bf16_t*)carve(16 * 32768 * 2);
  bf16_t* WPROJ = (bf16_t*)carve(4 * 16 * 512 * 2);
  bf16_t* WNOUT = (bf16_t*)carve(8 * 8 * 512 * 2);
  bf16_t* WEOUT = (bf16_t*)carve(8 * 4 * 512 * 2);
  for (int l = 0; l < 4; ++l) BN[l] = (float*)carve(1024 * 4);
  for (int l = 0; l < 4; ++l) BE[l] = (float*)carve(1024 * 4);
  bf16_t* zbf   = (bf16_t*)carve(B_ * 128 * 2);
  bf16_t* zc    = (bf16_t*)carve(BH * 2);
  bf16_t* embs  = (bf16_t*)carve((size_t)S_ * B_ * NE_ * 2);
  bf16_t* hN    = (bf16_t*)carve(4 * 2 * BH * 2);
  bf16_t* hE    = (bf16_t*)carve(4 * 2 * BH * 2);
  // zero block (contiguous): cN, cE, xnode
  char* zero_base = ws + off;
  float*  cN    = (float*)carve(4 * BH * 4);
  float*  cE    = (float*)carve(4 * BH * 4);
  bf16_t* xnode = (bf16_t*)carve(BH * 2);
  size_t zero_bytes = (size_t)((char*)(xnode) + ((BH * 2 + 255) & ~255) -
                               zero_base);
  hipMemsetAsync(zero_base, 0, zero_bytes, stream);

  // ---- pack weights ----
  for (int l = 0; l < 4; ++l) {
    int total = 16 * 32768;
    pack_lstm_w<<<total / 256, 256, 0, stream>>>(F(5 + 4 * l), F(6 + 4 * l),
                                                 WN[l], H_, total);
    pack_lstm_bias<<<4, 256, 0, stream>>>(F(7 + 4 * l), F(8 + 4 * l), BN[l]);
  }
  {
    int total = 12 * 32768;
    pack_lstm_w<<<total / 256, 256, 0, stream>>>(F(23), F(24), WE[0], NE_, total);
    pack_lstm_bias<<<4, 256, 0, stream>>>(F(25), F(26), BE[0]);
  }
  for (int l = 1; l < 4; ++l) {
    int total = 16 * 32768;
    pack_lstm_w<<<total / 256, 256, 0, stream>>>(F(23 + 4 * l), F(24 + 4 * l),
                                                 WE[l], H_, total);
    pack_lstm_bias<<<4, 256, 0, stream>>>(F(25 + 4 * l), F(26 + 4 * l), BE[l]);
  }
  pack_dense_w<<<(4 * 16 * 512) / 256, 256, 0, stream>>>(F(3), WPROJ, 16, 128,
                                                         4 * 16 * 512);
  pack_dense_w<<<(8 * 8 * 512) / 256, 256, 0, stream>>>(F(21), WNOUT, 8, H_,
                                                        8 * 8 * 512);
  pack_dense_w<<<(8 * 4 * 512) / 256, 256, 0, stream>>>(F(39), WEOUT, 4, H_,
                                                        8 * 4 * 512);
  cvt_f32_bf16<<<(B_ * 128) / 256, 256, 0, stream>>>(F(0), zbf, B_ * 128);

  // ---- main persistent kernel ----
  RP p;
  for (int l = 0; l < 4; ++l) { p.WN[l] = WN[l]; p.WE[l] = WE[l];
                                p.BN[l] = BN[l]; p.BE[l] = BE[l]; }
  p.WPROJ = WPROJ; p.WNOUT = WNOUT; p.WEOUT = WEOUT;
  p.proj_b = F(4); p.nout_b = F(22); p.eout_b = F(40);
  p.zbf = zbf; p.zc = zc; p.embs = embs; p.hN = hN; p.hE = hE;
  p.xnode = xnode; p.cN = cN; p.cE = cE;
  p.atom = I(1); p.bond = I(2);
  for (int j = 0; j < 4; ++j) {
    p.ncW[j] = F(41 + 2 * j); p.ncB[j] = F(42 + 2 * j);
    p.ecW[j] = F(49 + 2 * j); p.ecB[j] = F(50 + 2 * j);
  }
  p.out = (float*)d_out;

  rnn_main<<<1, 512, 0, stream>>>(p);
}
